// GatedResidualNetwork_67027259621479
// MI455X (gfx1250) — compile-verified, matrix operand order checked by reading, not run on hardware
//
#include <hip/hip_runtime.h>
#include <hip/hip_bf16.h>

// ---------------------------------------------------------------------------
// Types for WMMA fragments
// ---------------------------------------------------------------------------
typedef __attribute__((ext_vector_type(16))) __bf16 bf16x16;
typedef __attribute__((ext_vector_type(8)))  float  f32x8;

union Frag {
    uint4   u[2];   // two contiguous 16B chunks (8 bf16 each)
    bf16x16 v;
};

__device__ __forceinline__ unsigned short bf16rne(float f) {
    unsigned int u = __float_as_uint(f);
    u += 0x7FFFu + ((u >> 16) & 1u);       // round-to-nearest-even
    return (unsigned short)(u >> 16);
}

constexpr int BK  = 32;   // K chunk per LDS stage (one WMMA K)
constexpr int LDT = 48;   // LDS row stride in bf16 elements (96B, 16B aligned)

// ---------------------------------------------------------------------------
// gfx1250 async global -> LDS copy (16B per lane), tracked by ASYNCcnt.
// VDST = per-lane LDS byte offset; flat shared address low 32 bits are the
// LDS address per the aperture rule (ISA 00_overview 10.2).
// ---------------------------------------------------------------------------
__device__ __forceinline__ unsigned lds_addr32(const void* p) {
    return (unsigned)(unsigned long long)(uintptr_t)p;
}

__device__ __forceinline__ void async_copy_b128(unsigned lds_off, const void* gaddr) {
    asm volatile("global_load_async_to_lds_b128 %0, %1, off"
                 :: "v"(lds_off), "v"(gaddr)
                 : "memory");
}

__device__ __forceinline__ void wait_async0() {
    asm volatile("s_wait_asynccnt 0x0" ::: "memory");
}

// A/B 16-bit fragment (CDNA5 ISA 7.12.2): lane<16 -> K chunks {0..7,16..23},
// lane>=16 -> {8..15,24..31}; each chunk is 8 contiguous bf16 = one b128.
__device__ __forceinline__ void load_frag(const unsigned short* lds, int row, int lane, Frag& f) {
    const int kb = (lane >> 4) << 3;            // 0 or 8
    const unsigned short* p = lds + row * LDT + kb;
    f.u[0] = *(const uint4*)(p);
    f.u[1] = *(const uint4*)(p + 16);
}

__device__ __forceinline__ f32x8 wmma_bf16(const Frag& a, const Frag& b, f32x8 c) {
    return __builtin_amdgcn_wmma_f32_16x16x32_bf16(false, a.v, false, b.v,
                                                   (short)0, c, false, false);
}

// ---------------------------------------------------------------------------
// fp32 -> bf16 conversion of X (row-major, K contiguous)
// ---------------------------------------------------------------------------
__global__ __launch_bounds__(256) void convert_x_kernel(const float* __restrict__ X,
                                                        unsigned short* __restrict__ Xb) {
    const size_t i = ((size_t)blockIdx.x * 256 + threadIdx.x) * 4;
    const float4 v = *(const float4*)&X[i];
    uint2 o;
    o.x = (unsigned)bf16rne(v.x) | ((unsigned)bf16rne(v.y) << 16);
    o.y = (unsigned)bf16rne(v.z) | ((unsigned)bf16rne(v.w) << 16);
    *(uint2*)&Xb[i] = o;
}

// fp32 KxN weight -> bf16 NxK (transposed) so B-fragments are K-contiguous
__global__ __launch_bounds__(256) void convert_w_t_kernel(const float* __restrict__ W,
                                                          unsigned short* __restrict__ Wt) {
    __shared__ float t[32][33];
    const int nb = blockIdx.x * 32;
    const int kb = blockIdx.y * 32;
    const int tx = threadIdx.x & 31;
    const int ty = threadIdx.x >> 5;   // 0..7
#pragma unroll
    for (int r = ty; r < 32; r += 8)
        t[r][tx] = W[(size_t)(kb + r) * 1024 + nb + tx];
    __syncthreads();
#pragma unroll
    for (int r = ty; r < 32; r += 8)
        Wt[(size_t)(nb + r) * 1024 + kb + tx] = bf16rne(t[tx][r]);
}

// ---------------------------------------------------------------------------
// Tiled WMMA GEMM: C = A(bf16 MxK) * Bt(bf16 NxK)^T + bias
//   RELU_BF16_OUT=1: relu + bf16 store (feeds next GEMM)
//   RELU_BF16_OUT=0: plain f32 store
// Block tile 128x128x32, 8 waves, wave tile 32x64 (2x4 WMMA tiles).
// Double-buffered LDS with gfx1250 async global->LDS DMA: chunk k+1 streams
// into the alternate buffer while WMMAs consume chunk k; one barrier/iter.
// ---------------------------------------------------------------------------
template <int RELU_BF16_OUT>
__global__ __launch_bounds__(256) void gemm_bf16_kernel(
    const unsigned short* __restrict__ A,
    const unsigned short* __restrict__ Bt,
    const float* __restrict__ bias,
    void* __restrict__ Cout,
    int M, int N, int K)
{
    __shared__ unsigned short sA[2][128 * LDT];
    __shared__ unsigned short sB[2][128 * LDT];

    const int tid  = threadIdx.x;
    const int lane = tid & 31;
    const int wave = tid >> 5;
    const int wm   = (wave & 3) * 32;   // 4 waves along M
    const int wn   = (wave >> 2) * 64;  // 2 waves along N
    const size_t m0 = (size_t)blockIdx.y * 128;
    const size_t n0 = (size_t)blockIdx.x * 128;

    f32x8 acc[2][4] = {};

    // stage K-chunk k0 into buffer `buf` via async DMA (no VGPR round trip)
    auto stage = [&](int buf, int k0) {
#pragma unroll
        for (int i = tid; i < 128 * 4; i += 256) {
            const int r = i >> 2, q = (i & 3) << 3;
            async_copy_b128(lds_addr32(&sA[buf][r * LDT + q]),
                            &A[(m0 + r) * K + k0 + q]);
            async_copy_b128(lds_addr32(&sB[buf][r * LDT + q]),
                            &Bt[(n0 + r) * K + k0 + q]);
        }
    };

    stage(0, 0);
    wait_async0();
    __syncthreads();

    int cur = 0;
    for (int k0 = 0; k0 < K; k0 += BK) {
        if (k0 + BK < K) stage(cur ^ 1, k0 + BK);   // DMA overlaps compute below

        Frag a[2], b[4];
        load_frag(sA[cur], wm + 0,  lane, a[0]);
        load_frag(sA[cur], wm + 16, lane, a[1]);
#pragma unroll
        for (int t = 0; t < 4; ++t) load_frag(sB[cur], wn + t * 16, lane, b[t]);
#pragma unroll
        for (int mi = 0; mi < 2; ++mi)
#pragma unroll
            for (int ni = 0; ni < 4; ++ni)
                acc[mi][ni] = wmma_bf16(a[mi], b[ni], acc[mi][ni]);

        wait_async0();
        __syncthreads();
        cur ^= 1;
    }

    // Epilogue. C/D layout: VGPR i -> M = i + 8*(lane>=16), N = lane&15.
    const int lc = lane & 15;
    const int lh = (lane >> 4) << 3;
#pragma unroll
    for (int mi = 0; mi < 2; ++mi) {
#pragma unroll
        for (int ni = 0; ni < 4; ++ni) {
            const size_t nn = n0 + wn + ni * 16 + lc;
            const float bv = bias[nn];
#pragma unroll
            for (int i = 0; i < 8; ++i) {
                const size_t mm = m0 + wm + mi * 16 + lh + i;
                float v = acc[mi][ni][i] + bv;
                if constexpr (RELU_BF16_OUT) {
                    v = fmaxf(v, 0.0f);
                    ((unsigned short*)Cout)[mm * N + nn] = bf16rne(v);
                } else {
                    ((float*)Cout)[mm * N + nn] = v;
                }
            }
        }
    }
}

// ---------------------------------------------------------------------------
// Dual GEMM sharing the A tile: g = X*Wg+bg, s = X*Ws+bs,
// fused epilogue y = x * sigmoid(g) + s.
// Block tile 128x64x32, 8 waves, wave tile 32x32 (2x2 tiles per matrix).
// Double-buffered LDS with async DMA, one barrier per K chunk.
// ---------------------------------------------------------------------------
__global__ __launch_bounds__(256) void dual_gemm_gate_kernel(
    const unsigned short* __restrict__ A,    // Xbf, MxK
    const unsigned short* __restrict__ Bgt,  // Wg^T, NxK
    const unsigned short* __restrict__ Bst,  // Ws^T, NxK
    const float* __restrict__ bg,
    const float* __restrict__ bs,
    const float* __restrict__ X2,            // x = h*W2+b2, MxN f32
    float* __restrict__ Y,                   // output y, MxN f32
    int M, int N, int K)
{
    __shared__ unsigned short sA[2][128 * LDT];
    __shared__ unsigned short sG[2][64 * LDT];
    __shared__ unsigned short sS[2][64 * LDT];

    const int tid  = threadIdx.x;
    const int lane = tid & 31;
    const int wave = tid >> 5;
    const int wm   = (wave & 3) * 32;
    const int wn   = (wave >> 2) * 32;
    const size_t m0 = (size_t)blockIdx.y * 128;
    const size_t n0 = (size_t)blockIdx.x * 64;

    f32x8 ag[2][2] = {};
    f32x8 as[2][2] = {};

    auto stage = [&](int buf, int k0) {
#pragma unroll
        for (int i = tid; i < 128 * 4; i += 256) {
            const int r = i >> 2, q = (i & 3) << 3;
            async_copy_b128(lds_addr32(&sA[buf][r * LDT + q]),
                            &A[(m0 + r) * K + k0 + q]);
        }
        {
            const int r = tid >> 2, q = (tid & 3) << 3;   // 64 rows * 4 quads = 256
            async_copy_b128(lds_addr32(&sG[buf][r * LDT + q]),
                            &Bgt[(n0 + r) * K + k0 + q]);
            async_copy_b128(lds_addr32(&sS[buf][r * LDT + q]),
                            &Bst[(n0 + r) * K + k0 + q]);
        }
    };

    stage(0, 0);
    wait_async0();
    __syncthreads();

    int cur = 0;
    for (int k0 = 0; k0 < K; k0 += BK) {
        if (k0 + BK < K) stage(cur ^ 1, k0 + BK);

        Frag a[2], g[2], s[2];
        load_frag(sA[cur], wm + 0,  lane, a[0]);
        load_frag(sA[cur], wm + 16, lane, a[1]);
        load_frag(sG[cur], wn + 0,  lane, g[0]);
        load_frag(sG[cur], wn + 16, lane, g[1]);
        load_frag(sS[cur], wn + 0,  lane, s[0]);
        load_frag(sS[cur], wn + 16, lane, s[1]);
#pragma unroll
        for (int mi = 0; mi < 2; ++mi)
#pragma unroll
            for (int ni = 0; ni < 2; ++ni) {
                ag[mi][ni] = wmma_bf16(a[mi], g[ni], ag[mi][ni]);
                as[mi][ni] = wmma_bf16(a[mi], s[ni], as[mi][ni]);
            }

        wait_async0();
        __syncthreads();
        cur ^= 1;
    }

    const int lc = lane & 15;
    const int lh = (lane >> 4) << 3;
#pragma unroll
    for (int mi = 0; mi < 2; ++mi) {
#pragma unroll
        for (int ni = 0; ni < 2; ++ni) {
            const size_t nn = n0 + wn + ni * 16 + lc;
            const float gb = bg[nn];
            const float sb = bs[nn];
#pragma unroll
            for (int i = 0; i < 8; ++i) {
                const size_t mm  = m0 + wm + mi * 16 + lh + i;
                const size_t idx = mm * N + nn;
                const float gv  = ag[mi][ni][i] + gb;
                const float sv  = as[mi][ni][i] + sb;
                const float xv  = X2[idx];
                const float sig = 1.0f / (1.0f + __expf(-gv));
                Y[idx] = xv * sig + sv;
            }
        }
    }
}

// ---------------------------------------------------------------------------
// In-place row LayerNorm over 1024 elements (eps = 1e-3, Keras default)
// ---------------------------------------------------------------------------
__global__ __launch_bounds__(256) void layernorm_kernel(float* __restrict__ Y,
                                                        const float* __restrict__ gamma,
                                                        const float* __restrict__ beta) {
    __shared__ float s1[256];
    __shared__ float s2[256];
    const int tid = threadIdx.x;
    float* row = Y + (size_t)blockIdx.x * 1024;

    const float4 v = *(const float4*)&row[tid * 4];
    s1[tid] = v.x + v.y + v.z + v.w;
    s2[tid] = v.x * v.x + v.y * v.y + v.z * v.z + v.w * v.w;
    __syncthreads();
    for (int o = 128; o > 0; o >>= 1) {
        if (tid < o) { s1[tid] += s1[tid + o]; s2[tid] += s2[tid + o]; }
        __syncthreads();
    }
    const float mean = s1[0] * (1.0f / 1024.0f);
    const float var  = s2[0] * (1.0f / 1024.0f) - mean * mean;
    const float rstd = rsqrtf(var + 1e-3f);

    const float4 g = *(const float4*)&gamma[tid * 4];
    const float4 b = *(const float4*)&beta[tid * 4];
    float4 o;
    o.x = (v.x - mean) * rstd * g.x + b.x;
    o.y = (v.y - mean) * rstd * g.y + b.y;
    o.z = (v.z - mean) * rstd * g.z + b.z;
    o.w = (v.w - mean) * rstd * g.w + b.w;
    *(float4*)&row[tid * 4] = o;
}

// ---------------------------------------------------------------------------
// Launch
// ---------------------------------------------------------------------------
extern "C" void kernel_launch(void* const* d_in, const int* in_sizes, int n_in,
                              void* d_out, int out_size, void* d_ws, size_t ws_size,
                              hipStream_t stream) {
    const float* inp   = (const float*)d_in[0];
    const float* w1    = (const float*)d_in[1];
    const float* b1    = (const float*)d_in[2];
    const float* w2    = (const float*)d_in[3];
    const float* b2    = (const float*)d_in[4];
    const float* wg    = (const float*)d_in[5];
    const float* bg    = (const float*)d_in[6];
    const float* ws    = (const float*)d_in[7];
    const float* bs    = (const float*)d_in[8];
    const float* gamma = (const float*)d_in[9];
    const float* beta  = (const float*)d_in[10];

    const int D = 1024, U = 1024;
    const int M = in_sizes[0] / D;            // B*S = 16384

    // workspace layout
    char* base = (char*)d_ws;
    size_t off = 0;
    unsigned short* Xbf = (unsigned short*)(base + off); off += (size_t)M * D * 2;
    unsigned short* Hbf = (unsigned short*)(base + off); off += (size_t)M * U * 2;
    unsigned short* W1t = (unsigned short*)(base + off); off += (size_t)D * U * 2;
    unsigned short* W2t = (unsigned short*)(base + off); off += (size_t)U * U * 2;
    unsigned short* Wgt = (unsigned short*)(base + off); off += (size_t)D * U * 2;
    unsigned short* Wst = (unsigned short*)(base + off); off += (size_t)D * U * 2;
    float*          Xm  = (float*)(base + off);          off += (size_t)M * U * 4;
    (void)ws_size; (void)n_in; (void)out_size;

    // 1) conversions
    convert_x_kernel<<<(M * D) / 1024, 256, 0, stream>>>(inp, Xbf);
    dim3 tg(32, 32);
    convert_w_t_kernel<<<tg, 256, 0, stream>>>(w1, W1t);
    convert_w_t_kernel<<<tg, 256, 0, stream>>>(w2, W2t);
    convert_w_t_kernel<<<tg, 256, 0, stream>>>(wg, Wgt);
    convert_w_t_kernel<<<tg, 256, 0, stream>>>(ws, Wst);

    // 2) h = relu(X*W1 + b1)  -> bf16
    gemm_bf16_kernel<1><<<dim3(U / 128, M / 128), 256, 0, stream>>>(
        Xbf, W1t, b1, (void*)Hbf, M, U, D);

    // 3) x = h*W2 + b2 -> f32
    gemm_bf16_kernel<0><<<dim3(U / 128, M / 128), 256, 0, stream>>>(
        Hbf, W2t, b2, (void*)Xm, M, U, U);

    // 4) y = x*sigmoid(X*Wg+bg) + (X*Ws+bs) -> d_out
    dual_gemm_gate_kernel<<<dim3(U / 64, M / 128), 256, 0, stream>>>(
        Xbf, Wgt, Wst, bg, bs, Xm, (float*)d_out, M, U, D);

    // 5) LayerNorm in place on d_out
    layernorm_kernel<<<M, 256, 0, stream>>>((float*)d_out, gamma, beta);
}
